// TriangleMultiplication_47407849013494
// MI455X (gfx1250) — compile-verified, matrix operand order checked by reading, not run on hardware
//
#include <hip/hip_runtime.h>
#include <stdint.h>

#define N 512
#define CZ 128
#define CI 128
#define EPS 1e-5f

typedef __attribute__((ext_vector_type(16))) __bf16 bf16x16;
typedef __attribute__((ext_vector_type(8)))  float  f32x8;

union FragU { uint2 q[4]; bf16x16 v; };

__device__ __forceinline__ unsigned short f2bf(float f) {
  unsigned int u = __float_as_uint(f);
  u += 0x7fffu + ((u >> 16) & 1u);          // round-to-nearest-even
  return (unsigned short)(u >> 16);
}
__device__ __forceinline__ float bf2f(unsigned short h) {
  return __uint_as_float(((unsigned int)h) << 16);
}
__device__ __forceinline__ float sigmoidf(float x) {
  return 1.0f / (1.0f + __expf(-x));
}

// CDNA5 async global->LDS copy (per-lane addresses, tracked by ASYNCcnt).
__device__ __forceinline__ void async_ld_b128(unsigned lds_off, const void* g) {
  asm volatile("global_load_async_to_lds_b128 %0, %1, off"
               :: "v"(lds_off), "v"(g) : "memory");
}

// ---------------------------------------------------------------------------
// Kernel 1: LN(act) + 4 projections + gates.
// Block = 256 threads handles (row r, 32 consecutive s positions).
// Writes leftT/rightT channel-major [c][r][s] bf16 (via LDS transpose tile),
// gate natural [r][s][c] bf16.
// ---------------------------------------------------------------------------
__global__ __launch_bounds__(256) void k_ln_proj(
    const float* __restrict__ act, const float* __restrict__ mask,
    const float* __restrict__ ln_s, const float* __restrict__ ln_o,
    const float* __restrict__ w_left, const float* __restrict__ b_left,
    const float* __restrict__ w_right, const float* __restrict__ b_right,
    const float* __restrict__ w_lg, const float* __restrict__ b_lg,
    const float* __restrict__ w_rg, const float* __restrict__ b_rg,
    const float* __restrict__ w_gate, const float* __restrict__ b_gate,
    unsigned short* __restrict__ leftT, unsigned short* __restrict__ rightT,
    unsigned short* __restrict__ gate)
{
  __shared__ float xln[32][CZ];
  __shared__ float redS[32][8];
  __shared__ float redQ[32][8];
  __shared__ float stat[32][2];
  __shared__ __align__(16) unsigned short lt[CI][32];
  __shared__ __align__(16) unsigned short rt[CI][32];

  const int t = threadIdx.x;
  const int r = blockIdx.y;
  const int s0 = blockIdx.x * 32;

  const float4* base = (const float4*)(act + ((size_t)r * N + s0) * CZ);
#pragma unroll
  for (int it = 0; it < 4; ++it) {
    int idx = t + it * 256;          // 1024 float4 total
    float4 v = base[idx];
    int s = idx >> 5;                // 32 float4 per position
    int c = (idx & 31) * 4;
    *(float4*)&xln[s][c] = v;
  }
  __syncthreads();

  {
    const int s = t >> 3, g = t & 7;
    float sum = 0.f, sq = 0.f;
#pragma unroll
    for (int k = 0; k < 16; ++k) {
      float v = xln[s][g * 16 + k];
      sum += v; sq += v * v;
    }
    redS[s][g] = sum; redQ[s][g] = sq;
    __syncthreads();
    if (g == 0) {
      float ts = 0.f, tq = 0.f;
#pragma unroll
      for (int k = 0; k < 8; ++k) { ts += redS[s][k]; tq += redQ[s][k]; }
      float m = ts * (1.0f / CZ);
      float var = tq * (1.0f / CZ) - m * m;
      stat[s][0] = m;
      stat[s][1] = rsqrtf(var + EPS);
    }
    __syncthreads();
    float m = stat[s][0], rs = stat[s][1];
#pragma unroll
    for (int k = 0; k < 16; ++k) {
      int c = g * 16 + k;
      xln[s][c] = (xln[s][c] - m) * rs * ln_s[c] + ln_o[c];
    }
  }
  __syncthreads();

  {
    const int cout = t & 127;
    const int sp = t >> 7;
    const float bl = b_left[cout], br = b_right[cout];
    const float blg = b_lg[cout], brg = b_rg[cout], bg = b_gate[cout];
    for (int jj = 0; jj < 16; ++jj) {
      const int s = sp * 16 + jj;
      float al = 0.f, ar = 0.f, alg = 0.f, arg2 = 0.f, ag = 0.f;
#pragma unroll 4
      for (int cin = 0; cin < CZ; ++cin) {
        float xv = xln[s][cin];
        al   += xv * w_left [cin * CI + cout];
        alg  += xv * w_lg   [cin * CI + cout];
        ar   += xv * w_right[cin * CI + cout];
        arg2 += xv * w_rg   [cin * CI + cout];
        ag   += xv * w_gate [cin * CZ + cout];
      }
      float mk = mask[r * N + s0 + s];
      lt[cout][s] = f2bf(mk * (al + bl) * sigmoidf(alg + blg));
      rt[cout][s] = f2bf(mk * (ar + br) * sigmoidf(arg2 + brg));
      gate[((size_t)(r * N + s0 + s)) * CZ + cout] = f2bf(sigmoidf(ag + bg));
    }
  }
  __syncthreads();

  {
    const int c = t >> 1;
    const int so = (t & 1) * 16;
    const uint4* ls = (const uint4*)&lt[c][so];
    const uint4* rs2 = (const uint4*)&rt[c][so];
    size_t off = (size_t)c * N * N + (size_t)r * N + s0 + so;
    uint4* ld = (uint4*)(leftT + off);
    uint4* rd = (uint4*)(rightT + off);
    ld[0] = ls[0]; ld[1] = ls[1];
    rd[0] = rs2[0]; rd[1] = rs2[1];
  }
}

// ---------------------------------------------------------------------------
// Kernel 2: per-channel NT GEMM  y[c] = L[c] (512x512) * R[c]^T  in bf16 WMMA.
// WG = 256 threads (8 wave32), 128x256 output tile; wave = 64x64 (4x4 WMMA
// tiles). K steps of 32; double-buffered LDS fed by async global->LDS B128
// copies (ASYNCcnt), so HBM/L2 latency overlaps the WMMA stream.
// LDS row stride = 40 bf16 (80B): 16B-aligned rows, conflict-free frag reads.
// ---------------------------------------------------------------------------
__global__ __launch_bounds__(256) void k_tri_gemm(
    const unsigned short* __restrict__ leftT,
    const unsigned short* __restrict__ rightT,
    float* __restrict__ y)
{
  __shared__ __align__(16) unsigned short sA[2][128 * 40];
  __shared__ __align__(16) unsigned short sB[2][256 * 40];

  const int t = threadIdx.x;
  const int lane = t & 31;
  const int wave = t >> 5;
  const int gi0 = blockIdx.x * 128;
  const int gj0 = blockIdx.y * 256;
  const int c = blockIdx.z;

  const unsigned short* Ap = leftT  + (size_t)c * N * N + (size_t)gi0 * N;
  const unsigned short* Bp = rightT + (size_t)c * N * N + (size_t)gj0 * N;

  const int mbase = (wave >> 2) * 64;   // 2 wave rows
  const int nbase = (wave & 3) * 64;    // 4 wave cols

  const unsigned sAoff[2] = { (unsigned)(uintptr_t)&sA[0][0],
                              (unsigned)(uintptr_t)&sA[1][0] };
  const unsigned sBoff[2] = { (unsigned)(uintptr_t)&sB[0][0],
                              (unsigned)(uintptr_t)&sB[1][0] };

  // Per-thread slot for cooperative staging (same for every tile).
  const int arow = (t + 0) >> 2;                 // A: 512 uint4 -> 2 per thread
  const int arow2 = (t + 256) >> 2;
  const int akc = (t & 3) * 8;
  const unsigned aslot  = (unsigned)(arow  * 80 + akc * 2);
  const unsigned aslot2 = (unsigned)(arow2 * 80 + akc * 2);

  const f32x8 zero = {0.f, 0.f, 0.f, 0.f, 0.f, 0.f, 0.f, 0.f};
  f32x8 acc[4][4];
#pragma unroll
  for (int mt = 0; mt < 4; ++mt)
#pragma unroll
    for (int nt = 0; nt < 4; ++nt)
      acc[mt][nt] = zero;

  const int lrow = lane & 15;
  const int kc = (lane >> 4) * 8;   // fragment K-chunk base per lane half

  const int NK = N / 32;

  // Issue one tile's async copies (6 x B128 per thread: 2 for A, 4 for B).
  auto issue_tile = [&](int buf, int k0) {
    async_ld_b128(sAoff[buf] + aslot,  Ap + (size_t)arow  * N + k0 + akc);
    async_ld_b128(sAoff[buf] + aslot2, Ap + (size_t)arow2 * N + k0 + akc);
#pragma unroll
    for (int it = 0; it < 4; ++it) {
      int idx = t + it * 256;                    // B: 1024 uint4
      int row = idx >> 2;
      int kc8 = (idx & 3) * 8;
      async_ld_b128(sBoff[buf] + (unsigned)(row * 80 + kc8 * 2),
                    Bp + (size_t)row * N + k0 + kc8);
    }
  };

  issue_tile(0, 0);

  for (int ki = 0; ki < NK; ++ki) {
    const int b = ki & 1;
    if (ki + 1 < NK) {
      issue_tile(1 - b, (ki + 1) * 32);
      // In-order completion: oldest 6 (current tile) done, next 6 in flight.
      asm volatile("s_wait_asynccnt 0x6" ::: "memory");
    } else {
      asm volatile("s_wait_asynccnt 0x0" ::: "memory");
    }
    __syncthreads();

    FragU bf[4];
#pragma unroll
    for (int nt = 0; nt < 4; ++nt) {
      const unsigned short* p = &sB[b][(nbase + nt * 16 + lrow) * 40];
      bf[nt].q[0] = *(const uint2*)(p + kc);
      bf[nt].q[1] = *(const uint2*)(p + kc + 4);
      bf[nt].q[2] = *(const uint2*)(p + kc + 16);
      bf[nt].q[3] = *(const uint2*)(p + kc + 20);
    }
#pragma unroll
    for (int mt = 0; mt < 4; ++mt) {
      FragU af;
      const unsigned short* p = &sA[b][(mbase + mt * 16 + lrow) * 40];
      af.q[0] = *(const uint2*)(p + kc);
      af.q[1] = *(const uint2*)(p + kc + 4);
      af.q[2] = *(const uint2*)(p + kc + 16);
      af.q[3] = *(const uint2*)(p + kc + 20);
#pragma unroll
      for (int nt = 0; nt < 4; ++nt) {
        acc[mt][nt] = __builtin_amdgcn_wmma_f32_16x16x32_bf16(
            false, af.v, false, bf[nt].v, (short)0, acc[mt][nt], false, false);
      }
    }
    __syncthreads();   // all waves done reading buf b before it is refilled
  }

  // Store: C/D layout — VGPR r holds (M = r + 8*(lane>>4), N = lane&15).
  float* yp = y + (size_t)c * N * N;
#pragma unroll
  for (int mt = 0; mt < 4; ++mt)
#pragma unroll
    for (int nt = 0; nt < 4; ++nt)
#pragma unroll
      for (int rr = 0; rr < 8; ++rr) {
        int gm = gi0 + mbase + mt * 16 + rr + ((lane >> 4) * 8);
        int gn = gj0 + nbase + nt * 16 + (lane & 15);
        yp[(size_t)gm * N + gn] = acc[mt][nt][rr];
      }
}

// ---------------------------------------------------------------------------
// Kernel 3: LN over c of y[c][i][j] + output projection + gate, write [i][j][c].
// Block = 256 threads handles (row i, 32 consecutive j positions).
// ---------------------------------------------------------------------------
__global__ __launch_bounds__(256) void k_out(
    const float* __restrict__ y, const unsigned short* __restrict__ gate,
    const float* __restrict__ ln_s, const float* __restrict__ ln_o,
    const float* __restrict__ w_out, const float* __restrict__ b_out,
    float* __restrict__ out)
{
  __shared__ float yt[CI][33];
  __shared__ float redS[8][32];
  __shared__ float redQ[8][32];
  __shared__ float stat[2][32];

  const int t = threadIdx.x;
  const int i = blockIdx.y;
  const int j0 = blockIdx.x * 32;

#pragma unroll
  for (int it = 0; it < 4; ++it) {
    int idx = t + it * 256;          // 1024 float4: 8 per channel
    int c = idx >> 3;
    int p = (idx & 7) * 4;
    float4 v = *(const float4*)(y + ((size_t)c * N + i) * N + j0 + p);
    yt[c][p + 0] = v.x; yt[c][p + 1] = v.y; yt[c][p + 2] = v.z; yt[c][p + 3] = v.w;
  }
  __syncthreads();

  {
    const int j = t & 31, g = t >> 5;
    float sum = 0.f, sq = 0.f;
#pragma unroll
    for (int k = 0; k < 16; ++k) {
      float v = yt[g * 16 + k][j];
      sum += v; sq += v * v;
    }
    redS[g][j] = sum; redQ[g][j] = sq;
    __syncthreads();
    if (g == 0) {
      float ts = 0.f, tq = 0.f;
#pragma unroll
      for (int k = 0; k < 8; ++k) { ts += redS[k][j]; tq += redQ[k][j]; }
      float m = ts * (1.0f / CI);
      float var = tq * (1.0f / CI) - m * m;
      stat[0][j] = m;
      stat[1][j] = rsqrtf(var + EPS);
    }
    __syncthreads();
    float m = stat[0][j], rs = stat[1][j];
#pragma unroll
    for (int k = 0; k < 16; ++k) {
      int c = g * 16 + k;
      yt[c][j] = (yt[c][j] - m) * rs * ln_s[c] + ln_o[c];
    }
  }
  __syncthreads();

  {
    const int cout = t & 127;
    const int jp = t >> 7;
    const float bo = b_out[cout];
    for (int jj = 0; jj < 16; ++jj) {
      const int j = jp * 16 + jj;
      float acc = 0.f;
#pragma unroll 4
      for (int cin = 0; cin < CI; ++cin)
        acc += yt[cin][j] * w_out[cin * CZ + cout];
      size_t pos = ((size_t)(i * N + j0 + j)) * CZ + cout;
      out[pos] = (acc + bo) * bf2f(gate[pos]);
    }
  }
}

// ---------------------------------------------------------------------------
extern "C" void kernel_launch(void* const* d_in, const int* in_sizes, int n_in,
                              void* d_out, int out_size, void* d_ws, size_t ws_size,
                              hipStream_t stream)
{
  (void)in_sizes; (void)n_in; (void)out_size; (void)ws_size;
  const float* act          = (const float*)d_in[0];
  const float* mask         = (const float*)d_in[1];
  const float* ln_in_scale  = (const float*)d_in[2];
  const float* ln_in_offset = (const float*)d_in[3];
  const float* w_left  = (const float*)d_in[4];
  const float* b_left  = (const float*)d_in[5];
  const float* w_right = (const float*)d_in[6];
  const float* b_right = (const float*)d_in[7];
  const float* w_lg = (const float*)d_in[8];
  const float* b_lg = (const float*)d_in[9];
  const float* w_rg = (const float*)d_in[10];
  const float* b_rg = (const float*)d_in[11];
  const float* ln_c_scale  = (const float*)d_in[12];
  const float* ln_c_offset = (const float*)d_in[13];
  const float* w_out  = (const float*)d_in[14];
  const float* b_out  = (const float*)d_in[15];
  const float* w_gate = (const float*)d_in[16];
  const float* b_gate = (const float*)d_in[17];

  char* ws = (char*)d_ws;
  const size_t bf_plane = (size_t)N * N * CI * sizeof(unsigned short); // 64 MiB
  unsigned short* leftT  = (unsigned short*)(ws);
  unsigned short* rightT = (unsigned short*)(ws + bf_plane);
  unsigned short* gateb  = (unsigned short*)(ws + 2 * bf_plane);
  float* ybuf            = (float*)(ws + 3 * bf_plane);

  k_ln_proj<<<dim3(N / 32, N), 256, 0, stream>>>(
      act, mask, ln_in_scale, ln_in_offset,
      w_left, b_left, w_right, b_right, w_lg, b_lg, w_rg, b_rg,
      w_gate, b_gate, leftT, rightT, gateb);

  k_tri_gemm<<<dim3(N / 128, N / 256, CI), 256, 0, stream>>>(leftT, rightT, ybuf);

  k_out<<<dim3(N / 32, N), 256, 0, stream>>>(
      ybuf, gateb, ln_c_scale, ln_c_offset, w_out, b_out, (float*)d_out);
}
